// AtlasAttentionWrapper_26182120637008
// MI455X (gfx1250) — compile-verified, matrix operand order checked by reading, not run on hardware
//
#include <hip/hip_runtime.h>
#include <hip/hip_bf16.h>

// ---------------- types for WMMA operands (gfx1250, wave32) ----------------
typedef __bf16 v8bf  __attribute__((ext_vector_type(8)));
typedef __bf16 v16bf __attribute__((ext_vector_type(16)));
typedef float  v8f   __attribute__((ext_vector_type(8)));

#define BATCH  4
#define QL     1024
#define HIDDEN 4096
#define NH     32
#define NKV    8
#define HD     128
#define PAST   1024
#define SKV    (PAST + QL)          // 2048
#define MQ     (BATCH * QL)         // 4096
#define KVDIM  (NKV * HD)           // 1024
#define SOFT_SCALE 0.088388347648318447f   // 1/sqrt(128)

// GEMM tiling
#define BM   128
#define BN   128
#define BK   32
#define LDT  (BK + 8)               // padded LDS row stride (40 bf16 = 20 banks)

__device__ __forceinline__ unsigned short f32_bf16(float f) {
  union { float f; unsigned u; } x; x.f = f;
  unsigned r = x.u + 0x7FFFu + ((x.u >> 16) & 1u);   // round-to-nearest-even
  return (unsigned short)(r >> 16);
}

// 16x32 bf16 WMMA fragment from row-major storage with row stride ld.
// Lane L<16 : row/col L, K = {k0..k0+7, k0+16..k0+23}
// Lane L>=16: row/col L-16, K = {k0+8..k0+15, k0+24..k0+31}
__device__ __forceinline__ v16bf load_frag(const __bf16* __restrict__ base,
                                           int ld, int k0, int lane) {
  const int r  = lane & 15;
  const int hi = (lane >> 4) & 1;
  const __bf16* p = base + (size_t)r * ld + k0 + hi * 8;
  union { struct { v8bf lo, hi; } s; v16bf v; } u;
  u.s.lo = *(const v8bf*)(p);
  u.s.hi = *(const v8bf*)(p + 16);
  return u.v;
}

// same fragment pattern but from an LDS tile with stride LDT
__device__ __forceinline__ v16bf lds_frag(const __bf16* base, int lane) {
  const int r  = lane & 15;
  const int hi = (lane >> 4) & 1;
  const __bf16* p = base + r * LDT + hi * 8;
  union { struct { v8bf lo, hi; } s; v16bf v; } u;
  u.s.lo = *(const v8bf*)(p);
  u.s.hi = *(const v8bf*)(p + 16);
  return u.v;
}

// ---------------- fp32 -> bf16 conversion ----------------
__global__ void k_cvt_bf16(const float* __restrict__ in,
                           unsigned short* __restrict__ out, int n) {
  int i = blockIdx.x * blockDim.x + threadIdx.x;
  if (i < n) out[i] = f32_bf16(in[i]);
}

// ---------------- bf16 GEMM: out[M,N] = A[M,K] * W[N,K]^T + bias ----------------
// 128x128 block tile, LDS double-buffered, 8 waves x (32M x 64N) each.
// Per k-step: 16KB global traffic for 1 MFLOP -> ~64 FLOP/byte beyond LDS.
__global__ void __launch_bounds__(256)
k_gemm_bf16(const __bf16* __restrict__ A, const __bf16* __restrict__ W,
            const float* __restrict__ bias, void* __restrict__ out,
            int M, int N, int K, int out_bf16) {
  __shared__ __attribute__((aligned(16))) __bf16 sA[2][BM][LDT];
  __shared__ __attribute__((aligned(16))) __bf16 sB[2][BN][LDT];

  const int tid  = threadIdx.x;
  const int lane = tid & 31;
  const int wave = tid >> 5;
  const int wm = wave >> 1;                 // 0..3 : 32-row strip
  const int wn = wave & 1;                  // 0..1 : 64-col strip
  const int m0 = blockIdx.y * BM;
  const int n0 = blockIdx.x * BN;

  // staging map: thread t covers rows {t>>2, (t>>2)+64}, k-chunk (t&3)*8
  const int r0  = tid >> 2;                 // 0..63
  const int kof = (tid & 3) * 8;
  const __bf16* Ag = A + (size_t)m0 * K;
  const __bf16* Wg = W + (size_t)n0 * K;

  auto stage = [&](int buf, int k0) {
    v8bf a0 = *(const v8bf*)(Ag + (size_t)r0 * K        + k0 + kof);
    v8bf a1 = *(const v8bf*)(Ag + (size_t)(r0 + 64) * K + k0 + kof);
    v8bf b0 = *(const v8bf*)(Wg + (size_t)r0 * K        + k0 + kof);
    v8bf b1 = *(const v8bf*)(Wg + (size_t)(r0 + 64) * K + k0 + kof);
    *(v8bf*)&sA[buf][r0][kof]      = a0;
    *(v8bf*)&sA[buf][r0 + 64][kof] = a1;
    *(v8bf*)&sB[buf][r0][kof]      = b0;
    *(v8bf*)&sB[buf][r0 + 64][kof] = b1;
  };

  v8f acc[2][4];
  #pragma unroll
  for (int i = 0; i < 2; ++i)
    #pragma unroll
    for (int j = 0; j < 4; ++j) acc[i][j] = {};

  stage(0, 0);
  __syncthreads();

  const int nk = K / BK;
  for (int kt = 0; kt < nk; ++kt) {
    const int buf = kt & 1;
    if (kt + 1 < nk) stage(buf ^ 1, (kt + 1) * BK);   // prefetch overlaps WMMA

    v16bf af[2], bfr[4];
    #pragma unroll
    for (int i = 0; i < 2; ++i) af[i]  = lds_frag(&sA[buf][wm * 32 + i * 16][0], lane);
    #pragma unroll
    for (int j = 0; j < 4; ++j) bfr[j] = lds_frag(&sB[buf][wn * 64 + j * 16][0], lane);

    #pragma unroll
    for (int i = 0; i < 2; ++i)
      #pragma unroll
      for (int j = 0; j < 4; ++j)
        acc[i][j] = __builtin_amdgcn_wmma_f32_16x16x32_bf16(
            false, af[i], false, bfr[j], (short)0, acc[i][j], false, false);

    __syncthreads();
  }

  // C/D layout: lane<16 -> N=lane, M rows 0..7 ; lane>=16 -> N=lane-16, M rows 8..15
  const int nn = lane & 15;
  const int mb = (lane >> 4) * 8;
  #pragma unroll
  for (int i = 0; i < 2; ++i) {
    #pragma unroll
    for (int j = 0; j < 4; ++j) {
      const int mm = m0 + wm * 32 + i * 16 + mb;
      const int nc = n0 + wn * 64 + j * 16 + nn;
      const float bv = bias[nc];
      #pragma unroll
      for (int r = 0; r < 8; ++r) {
        float v = acc[i][j][r] + bv;
        if (out_bf16) ((unsigned short*)out)[(size_t)(mm + r) * N + nc] = f32_bf16(v);
        else          ((float*)out)[(size_t)(mm + r) * N + nc] = v;
      }
    }
  }
}

// ---------------- KV-cache concat (bf16) ----------------
// K_cat[b][hkv][s][d]
__global__ void k_concat_k(const float* __restrict__ past,
                           const unsigned short* __restrict__ knew,
                           unsigned short* __restrict__ out) {
  size_t i = (size_t)blockIdx.x * blockDim.x + threadIdx.x;
  if (i >= (size_t)BATCH * NKV * SKV * HD) return;
  int d   = (int)(i & (HD - 1));
  int s   = (int)((i >> 7) & (SKV - 1));
  int hkv = (int)((i >> 18) & (NKV - 1));
  int b   = (int)(i >> 21);
  unsigned short bits;
  if (s < PAST) bits = f32_bf16(past[(((size_t)(b * NKV + hkv)) * PAST + s) * HD + d]);
  else          bits = knew[((size_t)(b * QL + (s - PAST))) * KVDIM + hkv * HD + d];
  out[i] = bits;
}

// V^T[b][hkv][d][s] — transposed so the P*V B-operand is contiguous in s
__global__ void k_concat_vt(const float* __restrict__ past,
                            const unsigned short* __restrict__ vnew,
                            unsigned short* __restrict__ out) {
  size_t i = (size_t)blockIdx.x * blockDim.x + threadIdx.x;
  if (i >= (size_t)BATCH * NKV * HD * SKV) return;
  int s   = (int)(i & (SKV - 1));
  int d   = (int)((i >> 11) & (HD - 1));
  int hkv = (int)((i >> 18) & (NKV - 1));
  int b   = (int)(i >> 21);
  unsigned short bits;
  if (s < PAST) bits = f32_bf16(past[(((size_t)(b * NKV + hkv)) * PAST + s) * HD + d]);
  else          bits = vnew[((size_t)(b * QL + (s - PAST))) * KVDIM + hkv * HD + d];
  out[i] = bits;
}

// ---------------- fused SDPA: one WG per (b, h, 16 query rows) ----------------
// LDS: f32 scores [16][2048] (128KB) + bf16 probs [16][2048] (64KB) = 192KB,
// exploiting CDNA5's 320KB-per-WGP LDS (non-flash, zero global round trips).
__global__ void __launch_bounds__(256, 1)
k_attention(const __bf16* __restrict__ Qbf, const __bf16* __restrict__ Kcat,
            const __bf16* __restrict__ Vt, const int* __restrict__ amask,
            unsigned short* __restrict__ attn_out) {
  extern __shared__ char smem[];
  float*  s_sc = (float*)smem;                                       // [16][SKV]
  __bf16* s_p  = (__bf16*)(smem + (size_t)16 * SKV * sizeof(float)); // [16][SKV]

  const int lane = threadIdx.x & 31;
  const int wave = threadIdx.x >> 5;
  const int qt = blockIdx.x;           // 0..63
  const int h  = blockIdx.y;           // 0..31
  const int b  = blockIdx.z;           // 0..3
  const int hkv = h >> 2;              // GQA: groups of 4

  const int nn = lane & 15;
  const int mb = (lane >> 4) * 8;

  // Q fragments for this 16-row tile, reused across all 128 K tiles
  const __bf16* Qbase = Qbf + (size_t)(b * QL + qt * 16) * HIDDEN + h * HD;
  v16bf aq[4];
  #pragma unroll
  for (int dc = 0; dc < 4; ++dc) aq[dc] = load_frag(Qbase, HIDDEN, dc * 32, lane);

  // ---- stage 1: scores = (Q K^T) * 1/sqrt(d) into LDS ----
  const __bf16* Kh = Kcat + (size_t)(b * NKV + hkv) * SKV * HD;
  for (int st = wave; st < SKV / 16; st += 8) {
    v8f acc = {};
    const __bf16* Kbase = Kh + (size_t)st * 16 * HD;
    #pragma unroll
    for (int dc = 0; dc < 4; ++dc) {
      v16bf bk = load_frag(Kbase, HD, dc * 32, lane);
      acc = __builtin_amdgcn_wmma_f32_16x16x32_bf16(false, aq[dc], false, bk, (short)0, acc, false, false);
    }
    #pragma unroll
    for (int j = 0; j < 8; ++j)
      s_sc[(size_t)(mb + j) * SKV + st * 16 + nn] = acc[j] * SOFT_SCALE;
  }
  __syncthreads();

  // ---- stage 2: masked softmax, one wave per 2 rows ----
  const int* mrow = amask + b * SKV;
  #pragma unroll
  for (int rr = 0; rr < 2; ++rr) {
    const int row = wave * 2 + rr;
    float* srow = s_sc + (size_t)row * SKV;
    float mx = -3.0e38f;
    for (int c = lane; c < SKV; c += 32) {
      float v = srow[c];
      v = (mrow[c] == 0) ? v : -1.0e9f;
      srow[c] = v;
      mx = fmaxf(mx, v);
    }
    #pragma unroll
    for (int off = 16; off > 0; off >>= 1) mx = fmaxf(mx, __shfl_xor(mx, off, 32));
    float sum = 0.0f;
    for (int c = lane; c < SKV; c += 32) {
      float e = expf(srow[c] - mx);
      srow[c] = e;
      sum += e;
    }
    #pragma unroll
    for (int off = 16; off > 0; off >>= 1) sum += __shfl_xor(sum, off, 32);
    const float inv = 1.0f / sum;
    unsigned short* prow = (unsigned short*)(s_p + (size_t)row * SKV);
    for (int c = lane; c < SKV; c += 32) prow[c] = f32_bf16(srow[c] * inv);
  }
  __syncthreads();

  // ---- stage 3: out = P * V, one wave per 16-wide d tile ----
  const __bf16* Vbase = Vt + ((size_t)(b * NKV + hkv) * HD + wave * 16) * SKV;
  v8f acc = {};
  for (int s0 = 0; s0 < SKV; s0 += 32) {
    v16bf ap = load_frag(s_p, SKV, s0, lane);     // from LDS (ds_load_b128)
    v16bf bv = load_frag(Vbase, SKV, s0, lane);
    acc = __builtin_amdgcn_wmma_f32_16x16x32_bf16(false, ap, false, bv, (short)0, acc, false, false);
  }
  #pragma unroll
  for (int j = 0; j < 8; ++j) {
    size_t m = (size_t)(b * QL + qt * 16 + mb + j);
    attn_out[m * HIDDEN + h * HD + wave * 16 + nn] = f32_bf16(acc[j]);
  }
}

// ---------------- host ----------------
extern "C" void kernel_launch(void* const* d_in, const int* in_sizes, int n_in,
                              void* d_out, int out_size, void* d_ws, size_t ws_size,
                              hipStream_t stream) {
  const float* hs     = (const float*)d_in[0];
  const float* past_k = (const float*)d_in[1];
  const float* past_v = (const float*)d_in[2];
  const int*   amask  = (const int*)d_in[3];
  const float* Wq = (const float*)d_in[4];
  const float* bq = (const float*)d_in[5];
  const float* Wk = (const float*)d_in[6];
  const float* bk = (const float*)d_in[7];
  const float* Wv = (const float*)d_in[8];
  const float* bv = (const float*)d_in[9];
  const float* Wo = (const float*)d_in[10];
  const float* bo = (const float*)d_in[11];
  (void)in_sizes; (void)n_in; (void)out_size; (void)ws_size;

  char* w = (char*)d_ws;
  size_t off = 0;
  auto carve = [&](size_t bytes) {
    char* p = w + off;
    off += (bytes + 255) & ~(size_t)255;
    return p;
  };
  unsigned short* Xb   = (unsigned short*)carve((size_t)MQ * HIDDEN * 2);
  unsigned short* Wqb  = (unsigned short*)carve((size_t)HIDDEN * HIDDEN * 2);
  unsigned short* Wkb  = (unsigned short*)carve((size_t)KVDIM * HIDDEN * 2);
  unsigned short* Wvb  = (unsigned short*)carve((size_t)KVDIM * HIDDEN * 2);
  unsigned short* Wob  = (unsigned short*)carve((size_t)HIDDEN * HIDDEN * 2);
  unsigned short* Qb   = (unsigned short*)carve((size_t)MQ * HIDDEN * 2);
  unsigned short* Knew = (unsigned short*)carve((size_t)MQ * KVDIM * 2);
  unsigned short* Vnew = (unsigned short*)carve((size_t)MQ * KVDIM * 2);
  unsigned short* Kcat = (unsigned short*)carve((size_t)BATCH * NKV * SKV * HD * 2);
  unsigned short* Vt   = (unsigned short*)carve((size_t)BATCH * NKV * HD * SKV * 2);
  unsigned short* Attn = (unsigned short*)carve((size_t)MQ * HIDDEN * 2);

  const dim3 blk(256);
  auto cvt = [&](const float* in, unsigned short* out, size_t n) {
    k_cvt_bf16<<<dim3((unsigned)((n + 255) / 256)), blk, 0, stream>>>(in, out, (int)n);
  };
  cvt(hs, Xb,  (size_t)MQ * HIDDEN);
  cvt(Wq, Wqb, (size_t)HIDDEN * HIDDEN);
  cvt(Wk, Wkb, (size_t)KVDIM * HIDDEN);
  cvt(Wv, Wvb, (size_t)KVDIM * HIDDEN);
  cvt(Wo, Wob, (size_t)HIDDEN * HIDDEN);

  // QKV projections (bf16 WMMA, fp32 accumulate, LDS double-buffered tiles)
  k_gemm_bf16<<<dim3(HIDDEN / BN, MQ / BM), blk, 0, stream>>>(
      (const __bf16*)Xb, (const __bf16*)Wqb, bq, Qb, MQ, HIDDEN, HIDDEN, 1);
  k_gemm_bf16<<<dim3(KVDIM / BN, MQ / BM), blk, 0, stream>>>(
      (const __bf16*)Xb, (const __bf16*)Wkb, bk, Knew, MQ, KVDIM, HIDDEN, 1);
  k_gemm_bf16<<<dim3(KVDIM / BN, MQ / BM), blk, 0, stream>>>(
      (const __bf16*)Xb, (const __bf16*)Wvb, bv, Vnew, MQ, KVDIM, HIDDEN, 1);

  // KV-cache concat (+ V transpose)
  const size_t nkc = (size_t)BATCH * NKV * SKV * HD;
  k_concat_k<<<dim3((unsigned)((nkc + 255) / 256)), blk, 0, stream>>>(past_k, Knew, Kcat);
  k_concat_vt<<<dim3((unsigned)((nkc + 255) / 256)), blk, 0, stream>>>(past_v, Vnew, Vt);

  // fused SDPA with 192KB dynamic LDS per workgroup
  const size_t lds = (size_t)16 * SKV * sizeof(float) + (size_t)16 * SKV * sizeof(unsigned short);
  k_attention<<<dim3(QL / 16, NH, BATCH), blk, (unsigned)lds, stream>>>(
      (const __bf16*)Qb, (const __bf16*)Kcat, (const __bf16*)Vt, amask, Attn);

  // output projection -> fp32 result
  k_gemm_bf16<<<dim3(HIDDEN / BN, MQ / BM), blk, 0, stream>>>(
      (const __bf16*)Attn, (const __bf16*)Wob, bo, d_out, MQ, HIDDEN, HIDDEN, 0);
}
